// GATLayer_4355096838265
// MI455X (gfx1250) — matrix operand order repaired and checked by reading, not compiled, not run on hardware
//
#include <hip/hip_runtime.h>
#include <math.h>

// ---------------- problem constants (from reference) ----------------
#define DIM   128   // D
#define HC    128   // H*C
#define NH    8     // heads
#define CH    16    // per-head channels
#define EDF   16    // edge feature dim
#define NG    8     // graph segments
#define EPSX  1e-5f
#define SLOPE 0.2f

typedef __attribute__((ext_vector_type(16))) __bf16 v16bf;
typedef __attribute__((ext_vector_type(8)))  __bf16 v8bf;
typedef __attribute__((ext_vector_type(8)))  float  v8f;

// packed-W fragment count: 4 kc * 8 nt * 32 lanes * 16 bf16 = 16384 bf16 (32 KB)
#define WPK_ELEMS 16384

// ---------------- async global->LDS staging (gfx1250 ASYNCcnt path) ----------------
#if defined(__has_builtin)
#if __has_builtin(__builtin_amdgcn_global_load_async_to_lds_b128)
#define HAVE_ASYNC_LDS 1
#endif
#endif

#ifdef HAVE_ASYNC_LDS
typedef int vi4 __attribute__((vector_size(4 * sizeof(int))));
typedef __attribute__((address_space(1))) vi4 g_vi4;
typedef __attribute__((address_space(3))) vi4 l_vi4;
#endif

// copy 16 bytes global -> LDS (async DMA when available, else via VGPRs)
__device__ __forceinline__ void lds_copy16(void* lds, const void* g) {
#ifdef HAVE_ASYNC_LDS
  __builtin_amdgcn_global_load_async_to_lds_b128((g_vi4*)(void*)(size_t)g,
                                                 (l_vi4*)lds, 0, 0);
#else
  *(uint4*)lds = *(const uint4*)g;
#endif
}

__device__ __forceinline__ void lds_copy_join() {
#ifdef HAVE_ASYNC_LDS
#if __has_builtin(__builtin_amdgcn_s_wait_asynccnt)
  __builtin_amdgcn_s_wait_asynccnt(0);
#else
  asm volatile("s_wait_asynccnt 0" ::: "memory");
#endif
#endif
}

// ---------------- helpers ----------------
__device__ __forceinline__ void atomicMaxF32(float* addr, float val) {
  if (val >= 0.0f) atomicMax((int*)addr, __float_as_int(val));
  else             atomicMin((unsigned int*)addr, (unsigned int)__float_as_int(val));
}

__global__ void fill_f32(float* __restrict__ p, float v, long long n) {
  long long i = (long long)blockIdx.x * blockDim.x + threadIdx.x;
  if (i < n) p[i] = v;
}

// ---------------- 0a) convert x -> bf16 (row-major, once) ----------------
__global__ void cvt_bf16_kernel(const float* __restrict__ x, __bf16* __restrict__ xb,
                                long long n4) {
  long long i = (long long)blockIdx.x * blockDim.x + threadIdx.x;
  if (i >= n4) return;
  const float4 v = ((const float4*)x)[i];
  __bf16* o = xb + i * 4;
  o[0] = (__bf16)v.x; o[1] = (__bf16)v.y; o[2] = (__bf16)v.z; o[3] = (__bf16)v.w;
}

// ---------------- 0b) pre-swizzle W into WMMA B-fragment layout ----------------
// Wpk[((kc*8+nt)*32 + lane)*16 + j] = bf16( W[(kc*32 + (j<8 ? (lane>=16?8:0) + j : 16 + (lane>=16?8:0) + j - 8))*128 + nt*16 + (lane&15)] )
__global__ void pack_w_kernel(const float* __restrict__ W, __bf16* __restrict__ Wpk) {
  int i = blockIdx.x * blockDim.x + threadIdx.x;   // 1024 fragments-of-16
  if (i >= 1024) return;
  const int lane = i & 31, nt = (i >> 5) & 7, kc = i >> 8;
  const int kb2 = (lane >> 4) * 16;
  const int col = nt * 16 + (lane & 15);
  __bf16* o = Wpk + (size_t)i * 16;
#pragma unroll
  for (int j = 0; j < 16; ++j)
    o[j] = (__bf16)W[(size_t)(kc * 32 + (j < 8 ? kb2 / 2 + j : 16 + kb2 / 2 + j - 8)) * HC + col];
}

// ---------------- 1) degree + edge_attr sums for self-loop fill ----------------
__global__ void edge_stats_kernel(const int* __restrict__ dst,
                                  const float* __restrict__ edge_attr,
                                  float* __restrict__ deg,
                                  float* __restrict__ easum, long long E) {
  long long i = (long long)blockIdx.x * blockDim.x + threadIdx.x;
  if (i >= E * EDF) return;
  long long e = i >> 4; int j = (int)(i & 15);
  int d = dst[e];
  atomicAdd(easum + (size_t)d * EDF + j, edge_attr[e * EDF + j]);
  if (j == 0) atomicAdd(deg + d, 1.0f);
}

__global__ void ea_mean_kernel(float* __restrict__ easum,
                               const float* __restrict__ deg, long long n /*N*EDF*/) {
  long long i = (long long)blockIdx.x * blockDim.x + threadIdx.x;
  if (i >= n) return;
  easum[i] = easum[i] / fmaxf(deg[i >> 4], 1.0f);
}

// ---------------- 2) dual WMMA linear: xl = X@Wl+bl, xr = X@Wr+br ----------------
// One wave per 16-row tile; A loaded once; 2 outputs x 8 col-tiles x 4 K-chunks = 64 v_wmma.
// Both packed weight matrices staged in LDS (64 KB) per block via async global->LDS.
__global__ __launch_bounds__(256)
void wmma_dual_linear_kernel(const __bf16* __restrict__ Wlpk,
                             const __bf16* __restrict__ Wrpk,
                             const __bf16* __restrict__ Xbf,
                             const float* __restrict__ blv, const float* __restrict__ brv,
                             float* __restrict__ xl, float* __restrict__ xr, int nrows) {
  __shared__ __bf16 sW[2 * WPK_ELEMS];             // 64 KB
  {
    uint4* s4 = (uint4*)sW;
    const uint4* gl = (const uint4*)Wlpk;
    const uint4* gr = (const uint4*)Wrpk;
    for (int i = threadIdx.x; i < 2048; i += 256) {
      lds_copy16(s4 + i, gl + i);
      lds_copy16(s4 + 2048 + i, gr + i);
    }
    lds_copy_join();
  }
  __syncthreads();

  const int lane = threadIdx.x & 31;
  const int wave = threadIdx.x >> 5;
  const int tile = blockIdx.x * 8 + wave;
  if (tile * 16 >= nrows) return;                  // whole-wave exit (after barrier)
  const int hi  = lane >> 4;
  const int row = tile * 16 + (lane & 15);

  // A fragments (16x32 bf16 per chunk), pure b128 loads, no conversion
  v16bf a[4];
  const __bf16* xp = Xbf + (size_t)row * DIM;
#pragma unroll
  for (int kc = 0; kc < 4; ++kc) {
    const v8bf lo = *(const v8bf*)(xp + kc * 32 + hi * 8);
    const v8bf hv = *(const v8bf*)(xp + kc * 32 + 16 + hi * 8);
    a[kc] = __builtin_shufflevector(lo, hv, 0, 1, 2, 3, 4, 5, 6, 7,
                                            8, 9, 10, 11, 12, 13, 14, 15);
  }

#pragma unroll
  for (int nt = 0; nt < 8; ++nt) {
    const int col = nt * 16 + (lane & 15);
#pragma unroll
    for (int m = 0; m < 2; ++m) {
      v8f acc = {0.f, 0.f, 0.f, 0.f, 0.f, 0.f, 0.f, 0.f};
#pragma unroll
      for (int kc = 0; kc < 4; ++kc) {
        const v16bf bm = *(const v16bf*)(sW + m * WPK_ELEMS + ((kc * 8 + nt) * 32 + lane) * 16);
        acc = __builtin_amdgcn_wmma_f32_16x16x32_bf16(false, a[kc], false, bm,
                                                      (short)0, acc, false, false);
      }
      const float bb = (m ? brv : blv)[col];
      float* Out = m ? xr : xl;
#pragma unroll
      for (int i = 0; i < 8; ++i)
        Out[(size_t)(tile * 16 + i + hi * 8) * HC + col] = acc[i] + bb;
    }
  }
}

// ---------------- 3) per-edge attention logits (wave per edge) ----------------
__global__ __launch_bounds__(256)
void edge_logits_kernel(const float* __restrict__ We, const float* __restrict__ att,
                        const float* __restrict__ xl, const float* __restrict__ xr,
                        const float* __restrict__ edge_attr, const float* __restrict__ ea_mean,
                        const int* __restrict__ src, const int* __restrict__ dst,
                        float* __restrict__ elog, float* __restrict__ nmax,
                        long long E, long long E2) {
  __shared__ float sWe[EDF * HC];                  // 8 KB
  __shared__ float sAtt[HC];                       // 512 B
  {
    uint4* s4 = (uint4*)sWe;
    const uint4* g4 = (const uint4*)We;
    for (int i = threadIdx.x; i < (EDF * HC) / 4; i += 256) lds_copy16(s4 + i, g4 + i);
    if (threadIdx.x < HC / 4) lds_copy16((uint4*)sAtt + threadIdx.x,
                                         (const uint4*)att + threadIdx.x);
    lds_copy_join();
  }
  __syncthreads();

  const int lane = threadIdx.x & 31;
  const long long e = (long long)blockIdx.x * 8 + (threadIdx.x >> 5);
  if (e >= E2) return;

  int s, d; const float* ea;
  if (e < E) { s = src[e]; d = dst[e]; ea = edge_attr + e * EDF; }
  else       { s = (int)(e - E); d = s; ea = ea_mean + (size_t)s * EDF; }

  const int hc0 = lane * 4;
  const float4 vl = *(const float4*)(xl + (size_t)s * HC + hc0);
  const float4 vr = *(const float4*)(xr + (size_t)d * HC + hc0);
  float m0 = vl.x + vr.x, m1 = vl.y + vr.y, m2 = vl.z + vr.z, m3 = vl.w + vr.w;
#pragma unroll
  for (int j = 0; j < EDF; ++j) {
    const float aj = ea[j];
    const float4 w = *(const float4*)(sWe + j * HC + hc0);
    m0 += aj * w.x; m1 += aj * w.y; m2 += aj * w.z; m3 += aj * w.w;
  }
  m0 = m0 > 0.f ? m0 : SLOPE * m0;
  m1 = m1 > 0.f ? m1 : SLOPE * m1;
  m2 = m2 > 0.f ? m2 : SLOPE * m2;
  m3 = m3 > 0.f ? m3 : SLOPE * m3;
  float ep = m0 * sAtt[hc0] + m1 * sAtt[hc0 + 1] + m2 * sAtt[hc0 + 2] + m3 * sAtt[hc0 + 3];
  ep += __shfl_xor(ep, 1, 32);
  ep += __shfl_xor(ep, 2, 32);
  if ((lane & 3) == 0) {
    const int h = lane >> 2;
    elog[e * NH + h] = ep;
    atomicMaxF32(nmax + (size_t)d * NH + h, ep);
  }
}

// ---------------- 4) exp(e - max) + segment denominator ----------------
__global__ void edge_exp_kernel(const int* __restrict__ dst, float* __restrict__ elog,
                                const float* __restrict__ nmax, float* __restrict__ denom,
                                long long E, long long E2) {
  long long i = (long long)blockIdx.x * blockDim.x + threadIdx.x;
  if (i >= E2 * NH) return;
  long long e = i >> 3; int h = (int)(i & 7);
  int d = (e < E) ? dst[e] : (int)(e - E);
  float v = expf(elog[i] - nmax[(size_t)d * NH + h]);
  elog[i] = v;
  atomicAdd(denom + (size_t)d * NH + h, v);
}

// ---------------- 5) weighted scatter-aggregate (wave per edge) ----------------
__global__ __launch_bounds__(256)
void edge_agg_kernel(const float* __restrict__ xl, const int* __restrict__ src,
                     const int* __restrict__ dst, const float* __restrict__ elog,
                     const float* __restrict__ denom, float* __restrict__ agg,
                     long long E, long long E2) {
  const int lane = threadIdx.x & 31;
  const long long e = (long long)blockIdx.x * 8 + (threadIdx.x >> 5);
  if (e >= E2) return;
  int s, d;
  if (e < E) { s = src[e]; d = dst[e]; } else { s = (int)(e - E); d = s; }
  const int h = lane >> 2;
  const float alpha = elog[e * NH + h] / denom[(size_t)d * NH + h];
  const float4 v = *(const float4*)(xl + (size_t)s * HC + lane * 4);
  float* ap = agg + (size_t)d * HC + lane * 4;
  atomicAdd(ap + 0, alpha * v.x);
  atomicAdd(ap + 1, alpha * v.y);
  atomicAdd(ap + 2, alpha * v.z);
  atomicAdd(ap + 3, alpha * v.w);
}

// ---------------- 6) bias + residual, accumulate group sums ----------------
__global__ void post_kernel(const float* __restrict__ x, const float* __restrict__ bias,
                            const int* __restrict__ batch, float* __restrict__ agg,
                            float* __restrict__ gsum, float* __restrict__ gcnt,
                            long long n) {
  long long i = (long long)blockIdx.x * blockDim.x + threadIdx.x;
  if (i >= n) return;
  const int node = (int)(i >> 7), hc = (int)(i & 127);
  const float o = agg[i] + bias[hc] + x[i];
  agg[i] = o;
  const int g = batch[node];
  atomicAdd(gsum + g * HC + hc, o);
  if (hc == 0) atomicAdd(gcnt + g, 1.0f);
}

__global__ void group_mean_kernel(float* __restrict__ gsum, const float* __restrict__ gcnt) {
  int i = blockIdx.x * blockDim.x + threadIdx.x;
  if (i >= NG * HC) return;
  gsum[i] = gsum[i] / fmaxf(gcnt[i >> 7], 1.0f);    // becomes mean
}

__global__ void var_pass_kernel(const int* __restrict__ batch, const float* __restrict__ mean,
                                const float* __restrict__ gms, float* __restrict__ agg,
                                float* __restrict__ gvar, long long n) {
  long long i = (long long)blockIdx.x * blockDim.x + threadIdx.x;
  if (i >= n) return;
  const int node = (int)(i >> 7), hc = (int)(i & 127);
  const int g = batch[node];
  const float sub = agg[i] - gms[hc] * mean[g * HC + hc];
  agg[i] = sub;
  atomicAdd(gvar + g * HC + hc, sub * sub);
}

__global__ void group_rstd_kernel(float* __restrict__ gvar, const float* __restrict__ gcnt) {
  int i = blockIdx.x * blockDim.x + threadIdx.x;
  if (i >= NG * HC) return;
  const float var = gvar[i] / fmaxf(gcnt[i >> 7], 1.0f);
  gvar[i] = rsqrtf(var + EPSX);                     // becomes rstd
}

__global__ void final_kernel(const int* __restrict__ batch, const float* __restrict__ sub,
                             const float* __restrict__ rstd, const float* __restrict__ gw,
                             const float* __restrict__ gb, float* __restrict__ out,
                             long long n) {
  long long i = (long long)blockIdx.x * blockDim.x + threadIdx.x;
  if (i >= n) return;
  const int node = (int)(i >> 7), hc = (int)(i & 127);
  const int g = batch[node];
  const float o = gw[hc] * sub[i] * rstd[g * HC + hc] + gb[hc];
  out[i] = (o > 0.f) ? o : (expf(o) - 1.f);
}

// ---------------- launcher ----------------
static inline unsigned gridFor(long long n, int b) { return (unsigned)((n + b - 1) / b); }

extern "C" void kernel_launch(void* const* d_in, const int* in_sizes, int n_in,
                              void* d_out, int out_size, void* d_ws, size_t ws_size,
                              hipStream_t stream) {
  const float* x         = (const float*)d_in[0];
  const int*   edge_idx  = (const int*)d_in[1];
  const float* edge_attr = (const float*)d_in[2];
  const int*   batch     = (const int*)d_in[3];
  const float* Wl        = (const float*)d_in[4];
  const float* bl        = (const float*)d_in[5];
  const float* Wr        = (const float*)d_in[6];
  const float* br        = (const float*)d_in[7];
  const float* We        = (const float*)d_in[8];
  const float* att       = (const float*)d_in[9];
  const float* bias      = (const float*)d_in[10];
  const float* gn_w      = (const float*)d_in[11];
  const float* gn_b      = (const float*)d_in[12];
  const float* gn_ms     = (const float*)d_in[13];

  const long long N  = in_sizes[0] / DIM;
  const long long E  = in_sizes[2] / EDF;
  const long long E2 = E + N;
  const int* src = edge_idx;
  const int* dst = edge_idx + E;

  // workspace layout (float units)
  float* ws    = (float*)d_ws;
  float* xl    = ws;                     // N*128
  float* xr    = xl + N * HC;            // N*128
  float* elog  = xr + N * HC;            // E2*8
  float* zero0 = elog + E2 * NH;
  float* deg   = zero0;                  // N
  float* easum = deg + N;                // N*16 (-> ea_mean)
  float* denom = easum + N * EDF;        // N*8
  float* agg   = denom + N * NH;         // N*128
  float* gsum  = agg + N * HC;           // 1024 (-> mean)
  float* gvar  = gsum + NG * HC;         // 1024 (-> rstd)
  float* gcnt  = gvar + NG * HC;         // 8
  const long long zero_n = (gcnt + NG) - zero0;
  float* nmax  = gcnt + NG;              // N*8  (fill -inf)
  __bf16* xbf  = (__bf16*)(nmax + N * NH);  // N*128 bf16
  __bf16* wlpk = xbf + N * DIM;             // 16384 bf16 (32 KB)
  __bf16* wrpk = wlpk + WPK_ELEMS;          // 16384 bf16

  // init accumulators
  fill_f32<<<gridFor(zero_n, 256), 256, 0, stream>>>(zero0, 0.0f, zero_n);
  fill_f32<<<gridFor(N * NH, 256), 256, 0, stream>>>(nmax, -INFINITY, N * NH);

  // one-time conversions / packing
  cvt_bf16_kernel<<<gridFor(N * DIM / 4, 256), 256, 0, stream>>>(x, xbf, N * DIM / 4);
  pack_w_kernel<<<4, 256, 0, stream>>>(Wl, wlpk);
  pack_w_kernel<<<4, 256, 0, stream>>>(Wr, wrpk);

  // self-loop edge_attr mean
  edge_stats_kernel<<<gridFor(E * EDF, 256), 256, 0, stream>>>(dst, edge_attr, deg, easum, E);
  ea_mean_kernel<<<gridFor(N * EDF, 256), 256, 0, stream>>>(easum, deg, N * EDF);

  // dual dense transform via WMMA (bf16 in, f32 accumulate)
  const long long tiles = (N + 15) / 16;
  wmma_dual_linear_kernel<<<gridFor(tiles, 8), 256, 0, stream>>>(wlpk, wrpk, xbf, bl, br,
                                                                 xl, xr, (int)N);

  // attention
  edge_logits_kernel<<<gridFor(E2, 8), 256, 0, stream>>>(We, att, xl, xr, edge_attr, easum,
                                                         src, dst, elog, nmax, E, E2);
  edge_exp_kernel<<<gridFor(E2 * NH, 256), 256, 0, stream>>>(dst, elog, nmax, denom, E, E2);
  edge_agg_kernel<<<gridFor(E2, 8), 256, 0, stream>>>(xl, src, dst, elog, denom, agg, E, E2);

  // residual + GraphNorm + ELU
  post_kernel<<<gridFor(N * HC, 256), 256, 0, stream>>>(x, bias, batch, agg, gsum, gcnt, N * HC);
  group_mean_kernel<<<gridFor(NG * HC, 256), 256, 0, stream>>>(gsum, gcnt);
  var_pass_kernel<<<gridFor(N * HC, 256), 256, 0, stream>>>(batch, gsum, gn_ms, agg, gvar, N * HC);
  group_rstd_kernel<<<gridFor(NG * HC, 256), 256, 0, stream>>>(gvar, gcnt);
  final_kernel<<<gridFor(N * HC, 256), 256, 0, stream>>>(batch, agg, gvar, gn_w, gn_b,
                                                         (float*)d_out, N * HC);
}
